// SpectralAttention_10316511445328
// MI455X (gfx1250) — compile-verified, matrix operand order checked by reading, not run on hardware
//
#include <hip/hip_runtime.h>

// Problem constants (match reference)
#define BB   4
#define NN   65536
#define KK   32
#define C1V  64
#define EPSV 1e-5f

typedef float    v2f  __attribute__((ext_vector_type(2)));
typedef float    v8f  __attribute__((ext_vector_type(8)));
typedef _Float16 v16h __attribute__((ext_vector_type(16)));

// ---------------------------------------------------------------------------
// Kernel 1: q/k projection via WMMA.
// D = A(16x64) x B(64x16): A rows 0-2 = Wq, rows 3-5 = Wk, rows 6-15 = 0.
// One wave per 16-column tile of N. fp32 WMMA (16x16x4) x 16 K-steps.
// Output: q4[b*N+n] = (q0,q1,q2,0), k4[b*N+n] = (k0,k1,k2,0)  (16B gather granule)
// ---------------------------------------------------------------------------
__global__ __launch_bounds__(256) void qk_proj_kernel(
    const float* __restrict__ features,   // [B,C1,N]
    const float* __restrict__ Wq,         // [3,C1]
    const float* __restrict__ bq,         // [3]
    const float* __restrict__ Wk,         // [3,C1]
    const float* __restrict__ bk,         // [3]
    float4* __restrict__ q4,              // [B*N]
    float4* __restrict__ k4)              // [B*N]
{
    const int wave = blockIdx.x * (blockDim.x >> 5) + (threadIdx.x >> 5);
    const int lane = threadIdx.x & 31;
    const int row  = lane & 15;           // A: M index / B: N index
    const int hi   = lane >> 4;           // half-wave selects K pair
    const int tilesPerBatch = NN / 16;
    const int b  = wave / tilesPerBatch;
    const int n0 = (wave % tilesPerBatch) * 16;

    // A-matrix row source: rows 0-2 -> Wq, 3-5 -> Wk, 6-15 -> zero
    const float* wp = (row < 3) ? (Wq + row * C1V)
                    : (row < 6) ? (Wk + (row - 3) * C1V) : Wq;
    const float  wm = (row < 6) ? 1.0f : 0.0f;

    const float* fbase = features + (size_t)b * C1V * NN + (size_t)(n0 + row);

    v8f acc = {0.f,0.f,0.f,0.f,0.f,0.f,0.f,0.f};

#if __has_builtin(__builtin_amdgcn_wmma_f32_16x16x4_f32)
    #pragma unroll
    for (int s = 0; s < 16; ++s) {
        const int k0 = s * 4;
        const int kA = k0 + 2 * hi;       // ISA 7.12.2: vgpr0=K{0,2}, vgpr1=K{1,3}
        v2f a, bv;
        a[0]  = wm * wp[kA];
        a[1]  = wm * wp[kA + 1];
        bv[0] = fbase[(size_t)kA * NN];
        bv[1] = fbase[(size_t)(kA + 1) * NN];
        acc = __builtin_amdgcn_wmma_f32_16x16x4_f32(
                  false, a, false, bv, (short)0, acc, false, false);
    }
#else
    // Fallback: codegen-confirmed f16 WMMA 16x16x32, two K-halves of 32.
    #pragma unroll
    for (int halfk = 0; halfk < 2; ++halfk) {
        const int kbase = halfk * 32;
        v16h a, bv;
        #pragma unroll
        for (int v = 0; v < 8; ++v) {
            const int klo = (v < 4) ? (2 * v + 8 * hi)
                                    : (16 + 2 * (v - 4) + 8 * hi);
            a[2*v]    = (_Float16)(wm * wp[kbase + klo]);
            a[2*v+1]  = (_Float16)(wm * wp[kbase + klo + 1]);
            bv[2*v]   = (_Float16)fbase[(size_t)(kbase + klo) * NN];
            bv[2*v+1] = (_Float16)fbase[(size_t)(kbase + klo + 1) * NN];
        }
        acc = __builtin_amdgcn_wmma_f32_16x16x32_f16(
                  false, a, false, bv, (short)0, acc, false, false);
    }
#endif

    // D layout: lanes 0-15, acc[j] = row M=j, column N = n0+lane.
    if (lane < 16) {
        const size_t off = (size_t)b * NN + (n0 + row);
        q4[off] = make_float4(acc[0] + bq[0], acc[1] + bq[1], acc[2] + bq[2], 0.f);
        k4[off] = make_float4(acc[3] + bk[0], acc[4] + bk[1], acc[5] + bk[2], 0.f);
    }
}

// ---------------------------------------------------------------------------
// Kernel 2: fused gather + Gaussian kernels + folded MLP + softmax(K) + reduce.
// One wave per (b,n), one lane per neighbor k (K == 32 == wave32).
// Folded param table in LDS (80 floats):
//   P[0:6)  = bn0 scale      P[6:12)  = bn0 shift
//   P[12:48)= bn1_scale*W1   P[48:54) = bn1_scale*b1 + bn1_shift
//   P[54:72)= W2             P[72:75) = b2
// ---------------------------------------------------------------------------
__global__ __launch_bounds__(256) void attn_kernel(
    const float*  __restrict__ xyz,        // [B,N,3]
    const float*  __restrict__ intensity,  // [B,3,N]
    const int*    __restrict__ indices,    // [B,N,K]
    const float4* __restrict__ q4,         // [B*N]
    const float4* __restrict__ k4,         // [B*N]
    const float* __restrict__ bn0_g, const float* __restrict__ bn0_b,
    const float* __restrict__ bn0_m, const float* __restrict__ bn0_v,
    const float* __restrict__ W1,    const float* __restrict__ b1,
    const float* __restrict__ bn1_g, const float* __restrict__ bn1_b,
    const float* __restrict__ bn1_m, const float* __restrict__ bn1_v,
    const float* __restrict__ W2,    const float* __restrict__ b2,
    float* __restrict__ out)               // [B,3,N]
{
    __shared__ float P[80];
    const int t = threadIdx.x;

    if (t < 6) {
        const float s = bn0_g[t] * rsqrtf(bn0_v[t] + EPSV);
        P[t]     = s;
        P[6 + t] = bn0_b[t] - bn0_m[t] * s;
    } else if (t >= 32 && t < 68) {          // 36 entries of folded W1
        const int i = t - 32, o = i / 6;
        const float s1 = bn1_g[o] * rsqrtf(bn1_v[o] + EPSV);
        P[12 + i] = s1 * W1[i];
    } else if (t >= 68 && t < 74) {          // folded b1
        const int o = t - 68;
        const float s1 = bn1_g[o] * rsqrtf(bn1_v[o] + EPSV);
        P[48 + o] = s1 * b1[o] + (bn1_b[o] - bn1_m[o] * s1);
    } else if (t >= 96 && t < 114) {
        P[54 + (t - 96)] = W2[t - 96];
    } else if (t >= 114 && t < 117) {
        P[72 + (t - 114)] = b2[t - 114];
    }
    __syncthreads();

    const int wave = blockIdx.x * 8 + (t >> 5);
    const int lane = t & 31;
    const int b = wave >> 16;                // N == 65536
    const int n = wave & (NN - 1);
    const size_t bn = (size_t)b * NN + n;

    // Coalesced index load: 32 lanes read 32 consecutive ints (128B/wave).
    const int idx = indices[bn * KK + lane];
    const size_t gidx = (size_t)b * NN + idx;

    // Gathers (all L2-resident: per-batch sources are ~1MB each)
    const float4 kk = k4[gidx];
    const float4 qq = q4[bn];
    const float* xp = xyz + gidx * 3;
    const float xn0 = xp[0], xn1 = xp[1], xn2 = xp[2];
    const float* xc = xyz + bn * 3;
    const float cc0 = xc[0], cc1 = xc[1], cc2 = xc[2];
    const float* ib = intensity + (size_t)b * 3 * NN;
    float iv[3];
    iv[0] = ib[idx]; iv[1] = ib[NN + idx]; iv[2] = ib[2 * (size_t)NN + idx];

    // Gaussian kernels
    float h[6], d;
    d = kk.x - qq.x;  h[0] = __expf(-2.f * d * d);
    d = kk.y - qq.y;  h[1] = __expf(-2.f * d * d);
    d = kk.z - qq.z;  h[2] = __expf(-2.f * d * d);
    d = xn0 - cc0;    h[3] = __expf(-2.f * d * d);
    d = xn1 - cc1;    h[4] = __expf(-2.f * d * d);
    d = xn2 - cc2;    h[5] = __expf(-2.f * d * d);

    // BN0 + ReLU
    float hh[6];
    #pragma unroll
    for (int c = 0; c < 6; ++c)
        hh[c] = fmaxf(0.f, h[c] * P[c] + P[6 + c]);

    // (BN1 ∘ (W1·x + b1)) folded
    float g[6];
    #pragma unroll
    for (int o = 0; o < 6; ++o) {
        float s = P[48 + o];
        #pragma unroll
        for (int c = 0; c < 6; ++c) s += P[12 + o * 6 + c] * hh[c];
        g[o] = s;
    }

    // W2·g + b2
    float w[3];
    #pragma unroll
    for (int o = 0; o < 3; ++o) {
        float s = P[72 + o];
        #pragma unroll
        for (int c = 0; c < 6; ++c) s += P[54 + o * 6 + c] * g[c];
        w[o] = s;
    }

    // Per-channel softmax over K (== over lanes) fused with weighted sum.
    float r[3];
    #pragma unroll
    for (int c = 0; c < 3; ++c) {
        float m = w[c];
        #pragma unroll
        for (int off = 16; off; off >>= 1)
            m = fmaxf(m, __shfl_xor(m, off));
        const float e = __expf(w[c] - m);
        float se = e;
        float si = iv[c] * e;
        #pragma unroll
        for (int off = 16; off; off >>= 1) {
            se += __shfl_xor(se, off);
            si += __shfl_xor(si, off);
        }
        r[c] = si / se;
    }

    if      (lane == 0) out[((size_t)b * 3 + 0) * NN + n] = r[0];
    else if (lane == 1) out[((size_t)b * 3 + 1) * NN + n] = r[1];
    else if (lane == 2) out[((size_t)b * 3 + 2) * NN + n] = r[2];
}

// ---------------------------------------------------------------------------
extern "C" void kernel_launch(void* const* d_in, const int* in_sizes, int n_in,
                              void* d_out, int out_size, void* d_ws, size_t ws_size,
                              hipStream_t stream) {
    const float* xyz       = (const float*)d_in[0];
    const float* features  = (const float*)d_in[1];
    const float* intensity = (const float*)d_in[2];
    const int*   indices   = (const int*)  d_in[3];
    const float* Wq        = (const float*)d_in[4];
    const float* bq        = (const float*)d_in[5];
    const float* Wk        = (const float*)d_in[6];
    const float* bk        = (const float*)d_in[7];
    const float* bn0_g     = (const float*)d_in[8];
    const float* bn0_b     = (const float*)d_in[9];
    const float* bn0_m     = (const float*)d_in[10];
    const float* bn0_v     = (const float*)d_in[11];
    const float* W1        = (const float*)d_in[12];
    const float* b1        = (const float*)d_in[13];
    const float* bn1_g     = (const float*)d_in[14];
    const float* bn1_b     = (const float*)d_in[15];
    const float* bn1_m     = (const float*)d_in[16];
    const float* bn1_v     = (const float*)d_in[17];
    const float* W2        = (const float*)d_in[18];
    const float* b2        = (const float*)d_in[19];
    float* out = (float*)d_out;

    // Workspace: q4 then k4, each B*N float4 (4 MB each; 8 MB total).
    float4* q4 = (float4*)d_ws;
    float4* k4 = q4 + (size_t)BB * NN;

    // Kernel 1: B*N/16 waves, 8 waves (256 threads) per block.
    const int projBlocks = (BB * NN / 16) / 8;     // 2048
    qk_proj_kernel<<<projBlocks, 256, 0, stream>>>(features, Wq, bq, Wk, bk, q4, k4);

    // Kernel 2: one wave per (b,n): B*N waves, 8 per block.
    const int attnBlocks = (BB * NN) / 8;          // 32768
    attn_kernel<<<attnBlocks, 256, 0, stream>>>(
        xyz, intensity, indices, q4, k4,
        bn0_g, bn0_b, bn0_m, bn0_v, W1, b1,
        bn1_g, bn1_b, bn1_m, bn1_v, W2, b2, out);
}
